// SymbolicCodecV3_47545287967484
// MI455X (gfx1250) — compile-verified, matrix operand order checked by reading, not run on hardware
//
#include <hip/hip_runtime.h>
#include <stdint.h>

#define DEV __device__ __forceinline__

typedef __bf16 bf16_t;
typedef __attribute__((ext_vector_type(16))) __bf16 v16bf;
typedef __attribute__((ext_vector_type(8)))  float  v8f;
typedef __attribute__((ext_vector_type(8)))  unsigned int v8u;
typedef int gv4i __attribute__((vector_size(16)));   // matches builtin's V4i prototype

// ---------------- model dims ----------------
#define Bq    16
#define Pq    512
#define Sq    16
#define Eq    256
#define Hq    8
#define DHq   32
#define Lq    4
#define DLATq 256
#define Cq    4
#define KCBq  1024
#define DSq   64
#define Vq    259
#define PADq  258
#define MTOK  (Bq*Pq*Sq)   /* 131072 tokens */
#define BPq   (Bq*Pq)      /* 8192 patches */
#define M2q   (Bq*(Pq-1))  /* 8176 prior rows */

// ---------------- bf16 helpers (bit-exact, no __bf16 arithmetic) ----------------
DEV float bf2f(bf16_t b) {
    unsigned short s = __builtin_bit_cast(unsigned short, b);
    unsigned u = ((unsigned)s) << 16;
    return __builtin_bit_cast(float, u);
}
DEV bf16_t f2bf(float f) {
    unsigned u = __builtin_bit_cast(unsigned, f);
    unsigned r = u + 0x7fffu + ((u >> 16) & 1u);
    unsigned short s = (unsigned short)(r >> 16);
    return __builtin_bit_cast(bf16_t, s);
}
DEV v16bf vz16() {
    v8u z = {0u,0u,0u,0u,0u,0u,0u,0u};
    return __builtin_bit_cast(v16bf, z);
}
DEV v8f vz8f() {
    v8f z = {0.f,0.f,0.f,0.f,0.f,0.f,0.f,0.f};
    return z;
}

// ---------------- CDNA5 WMMA ----------------
DEV v8f wmma_bf16(v16bf a, v16bf b, v8f c) {
    // D(f32 16x16) = A(16x32 bf16) * B(32x16 bf16) + C
    return __builtin_amdgcn_wmma_f32_16x16x32_bf16(false, a, false, b, (short)0, c, false, false);
}

// ---------------- async global->LDS (guarded: compiles on both toolchains) ----------------
#ifndef USE_ASYNC
#if defined(__has_builtin)
#if __has_builtin(__builtin_amdgcn_global_load_async_to_lds_b128) && __has_builtin(__builtin_amdgcn_s_wait_asynccnt)
#define USE_ASYNC 1
#endif
#endif
#endif
#ifndef USE_ASYNC
#define USE_ASYNC 0
#endif

DEV void async_copy_b128(const void* g, void* l) {
#if USE_ASYNC
    __builtin_amdgcn_global_load_async_to_lds_b128(
        (__attribute__((address_space(1))) gv4i*)(unsigned long long)g,
        (__attribute__((address_space(3))) gv4i*)(unsigned int)(unsigned long long)l,
        0, 0);
#else
    (void)g; (void)l;
#endif
}
DEV void wait_async0() {
#if USE_ASYNC
    __builtin_amdgcn_s_wait_asynccnt(0);
#endif
}

// ---------------- block reductions (blockDim.x == 256) ----------------
DEV float blockReduceSum(float v, float* sh) {
    int t = threadIdx.x;
    sh[t] = v; __syncthreads();
    for (int o = 128; o > 0; o >>= 1) { if (t < o) sh[t] += sh[t + o]; __syncthreads(); }
    float r = sh[0]; __syncthreads();
    return r;
}
DEV float blockReduceMax(float v, float* sh) {
    int t = threadIdx.x;
    sh[t] = v; __syncthreads();
    for (int o = 128; o > 0; o >>= 1) { if (t < o) sh[t] = fmaxf(sh[t], sh[t + o]); __syncthreads(); }
    float r = sh[0]; __syncthreads();
    return r;
}

DEV float gelu_f(float x) { return 0.5f * x * (1.0f + erff(x * 0.70710678118f)); }

// =====================================================================
// WMMA GEMM: C[M,N] = act(A[M,K] @ W[N,K]^T + bias)
// A,W bf16 row-major; optional bf16 and/or f32 outputs.
// BM=128, BN=64, BK=32; 256 threads = 8 waves; wave tile 32x32 (2x2 WMMA).
// =====================================================================
#define BM 128
#define BN 64
#define BK 32

template<int ACT>
__global__ __launch_bounds__(256) void k_gemm(
    const bf16_t* __restrict__ A, const bf16_t* __restrict__ W,
    const float* __restrict__ bias,
    bf16_t* __restrict__ Cb, float* __restrict__ Cf,
    int M, int N, int K)
{
    __shared__ __align__(16) bf16_t As[2][BM][BK];
    __shared__ __align__(16) bf16_t Bs[2][BN][BK];

    const int tid  = threadIdx.x;
    const int lane = tid & 31;
    const int wave = tid >> 5;
    const int wM   = wave >> 1;   // 0..3
    const int wN   = wave & 1;    // 0..1
    const int m0   = blockIdx.y * BM;
    const int n0   = blockIdx.x * BN;

    auto loadStage = [&](int buf, int kt) {
        const int kk0 = kt * BK;
#if USE_ASYNC
        #pragma unroll
        for (int i = 0; i < 2; ++i) {               // A: 512 16B chunks, 2/thread
            int ca = tid * 2 + i;
            int r = ca >> 2, kc = ca & 3;
            int gr = m0 + r; gr = (gr < M) ? gr : (M - 1);
            async_copy_b128(&A[(size_t)gr * K + kk0 + kc * 8], &As[buf][r][kc * 8]);
        }
        {                                           // B: 256 16B chunks, 1/thread
            int r = tid >> 2, kc = tid & 3;
            int gr = n0 + r; gr = (gr < N) ? gr : (N - 1);
            async_copy_b128(&W[(size_t)gr * K + kk0 + kc * 8], &Bs[buf][r][kc * 8]);
        }
#else
        #pragma unroll
        for (int i = 0; i < 2; ++i) {
            int ca = tid * 2 + i;
            int r = ca >> 2, kc = ca & 3;
            int gr = m0 + r;
            uint4 vv = make_uint4(0u, 0u, 0u, 0u);
            if (gr < M) vv = *(const uint4*)&A[(size_t)gr * K + kk0 + kc * 8];
            *(uint4*)&As[buf][r][kc * 8] = vv;
        }
        {
            int r = tid >> 2, kc = tid & 3;
            int gr = n0 + r;
            uint4 vv = make_uint4(0u, 0u, 0u, 0u);
            if (gr < N) vv = *(const uint4*)&W[(size_t)gr * K + kk0 + kc * 8];
            *(uint4*)&Bs[buf][r][kc * 8] = vv;
        }
#endif
    };

    v8f acc[2][2];
    v8f zacc = vz8f();
    #pragma unroll
    for (int i = 0; i < 2; ++i)
        #pragma unroll
        for (int j = 0; j < 2; ++j) acc[i][j] = zacc;

    const int KT = K / BK;
    loadStage(0, 0);

    const int mlane = lane & 15;          // fragment row/col within 16
    const int khalf = (lane >> 4) * 16;   // K offset: lanes 0-15 -> K 0..15, 16-31 -> 16..31

    for (int kt = 0; kt < KT; ++kt) {
        const int buf = kt & 1;
        wait_async0();
        __syncthreads();
        if (kt + 1 < KT) loadStage(buf ^ 1, kt + 1);
        if (kt + 2 < KT) {
            __builtin_prefetch(&A[(size_t)(m0 + (tid & 127)) * K + (kt + 2) * BK], 0, 1);
            __builtin_prefetch(&W[(size_t)(n0 + (tid & 63))  * K + (kt + 2) * BK], 0, 1);
        }
        // A fragments: lane holds row m, 16 contiguous bf16 at its K-half (contiguous k order, same as B)
        v16bf a0 = *(const v16bf*)&As[buf][wM * 32 +  0 + mlane][khalf];
        v16bf a1 = *(const v16bf*)&As[buf][wM * 32 + 16 + mlane][khalf];
        // B fragments: lane holds col n, 16 contiguous bf16 of W row n at its K-half
        v16bf b0 = *(const v16bf*)&Bs[buf][wN * 32 +  0 + mlane][khalf];
        v16bf b1 = *(const v16bf*)&Bs[buf][wN * 32 + 16 + mlane][khalf];
        acc[0][0] = wmma_bf16(a0, b0, acc[0][0]);
        acc[0][1] = wmma_bf16(a0, b1, acc[0][1]);
        acc[1][0] = wmma_bf16(a1, b0, acc[1][0]);
        acc[1][1] = wmma_bf16(a1, b1, acc[1][1]);
        __syncthreads();
    }

    // epilogue: C layout — VGPR r: lanes 0-15 => M=r, lanes 16-31 => M=r+8; N = lane&15
    const int hi8 = (lane >> 4) * 8;
    #pragma unroll
    for (int mt = 0; mt < 2; ++mt) {
        #pragma unroll
        for (int nt = 0; nt < 2; ++nt) {
            const int col = n0 + wN * 32 + nt * 16 + mlane;
            const float bv = (bias && col < N) ? bias[col] : 0.0f;
            #pragma unroll
            for (int r = 0; r < 8; ++r) {
                const int row = m0 + wM * 32 + mt * 16 + hi8 + r;
                if (row < M && col < N) {
                    float v = acc[mt][nt][r] + bv;
                    if (ACT == 1)      v = v > 0.f ? v : 0.f;
                    else if (ACT == 2) v = gelu_f(v);
                    if (Cb) Cb[(size_t)row * N + col] = f2bf(v);
                    if (Cf) Cf[(size_t)row * N + col] = v;
                }
            }
        }
    }
}

// =====================================================================
// Attention, one (sequence, head) per 32-thread block; S=16, dh=32.
// scores = WMMA(q, k); softmax in-wave; out = WMMA(probs, v) x2 halves.
// =====================================================================
__global__ __launch_bounds__(32) void k_attn(
    const bf16_t* __restrict__ qkv, const unsigned char* __restrict__ padf,
    bf16_t* __restrict__ out, int useMask)
{
    const int b    = blockIdx.x / Hq;
    const int h    = blockIdx.x % Hq;
    const int lane = threadIdx.x;
    const int mrow = lane & 15;
    const int kh   = lane >> 4;
    const size_t rs = (size_t)3 * Eq;

    const bf16_t* qp = qkv + ((size_t)(b * Sq + mrow)) * rs + h * DHq + kh * 16;
    v16bf qa = *(const v16bf*)qp;            // A: q row mrow, dh-half kh
    v16bf ka = *(const v16bf*)(qp + Eq);     // B: k row mrow (column of k^T)
    v8f sc = vz8f();
    sc = wmma_bf16(qa, ka, sc);              // scores 16x16

    const float scale = 0.17677669529663687f; // 1/sqrt(32)
    float maskadd = 0.0f;
    if (useMask && padf[b * Sq + mrow]) maskadd = -1e9f;

    float p[8];
    #pragma unroll
    for (int r = 0; r < 8; ++r) {
        float v = sc[r] * scale + maskadd;
        float m = v;
        for (int d = 1; d < 16; d <<= 1) m = fmaxf(m, __shfl_xor(m, d, 32));
        float e = __expf(v - m);
        float s = e;
        for (int d = 1; d < 16; d <<= 1) s += __shfl_xor(s, d, 32);
        p[r] = e / s;
    }

    __shared__ bf16_t probs[16 * 16];
    #pragma unroll
    for (int r = 0; r < 8; ++r) probs[(r + kh * 8) * 16 + mrow] = f2bf(p[r]);
    __syncthreads();

    // probs as A fragment (K padded 16->32 with zeros in lanes 16-31)
    v16bf ap = vz16();
    if (kh == 0) ap = *(const v16bf*)&probs[mrow * 16];

    const bf16_t* vbase = qkv + 2 * Eq + h * DHq;
    #pragma unroll
    for (int half = 0; half < 2; ++half) {
        v16bf bv = vz16();
        if (kh == 0) {
            #pragma unroll
            for (int kk = 0; kk < 16; ++kk)
                bv[kk] = vbase[((size_t)(b * Sq + kk)) * rs + half * 16 + mrow];
        }
        v8f o = vz8f();
        o = wmma_bf16(ap, bv, o);
        #pragma unroll
        for (int r = 0; r < 8; ++r)
            out[((size_t)(b * Sq + r + kh * 8)) * Eq + h * DHq + half * 16 + mrow] = f2bf(o[r]);
    }
}

// =====================================================================
// small kernels
// =====================================================================
__global__ void k_cast_bf16(const float* __restrict__ src, bf16_t* __restrict__ dst, size_t n) {
    size_t i  = (size_t)blockIdx.x * 256 + threadIdx.x;
    size_t st = (size_t)gridDim.x * 256;
    for (; i < n; i += st) dst[i] = f2bf(src[i]);
}

__global__ void k_embed(const int* __restrict__ pid, const float* __restrict__ emb,
                        const float* __restrict__ pos, bf16_t* __restrict__ x,
                        unsigned char* __restrict__ padf) {
    size_t i = (size_t)blockIdx.x * 256 + threadIdx.x;   // over MTOK*E
    size_t token = i >> 8;
    int e = (int)(i & 255);
    int s = (int)(token & 15);
    int id = pid[token];
    x[i] = f2bf(emb[(size_t)id * Eq + e] + pos[s * Eq + e]);
    if (e == 0) padf[token] = (id == PADq) ? 1 : 0;
}

__global__ void k_addpos(const bf16_t* __restrict__ in, const float* __restrict__ pos,
                         bf16_t* __restrict__ out) {
    size_t i = (size_t)blockIdx.x * 256 + threadIdx.x;
    size_t token = i >> 8;
    int e = (int)(i & 255);
    int s = (int)(token & 15);
    out[i] = f2bf(bf2f(in[i]) + pos[s * Eq + e]);
}

__global__ __launch_bounds__(256) void k_ln256(
    const bf16_t* __restrict__ a, const bf16_t* __restrict__ res,
    const float* __restrict__ g, const float* __restrict__ b, bf16_t* __restrict__ out) {
    __shared__ float sh[256];
    const size_t row = blockIdx.x;
    const int t = threadIdx.x;
    float v = bf2f(a[row * 256 + t]) + (res ? bf2f(res[row * 256 + t]) : 0.0f);
    float mean = blockReduceSum(v, sh) * (1.0f / 256.0f);
    float d = v - mean;
    float var = blockReduceSum(d * d, sh) * (1.0f / 256.0f);
    out[row * 256 + t] = f2bf(d * rsqrtf(var + 1e-5f) * g[t] + b[t]);
}

__global__ __launch_bounds__(256) void k_maskln(
    const bf16_t* __restrict__ x, const unsigned char* __restrict__ padf,
    const float* __restrict__ g, const float* __restrict__ b, bf16_t* __restrict__ out) {
    __shared__ float sh[256];
    const int bp = blockIdx.x;
    const int t  = threadIdx.x;
    float loc[16];
    float s1 = 0.f, s2 = 0.f;
    #pragma unroll
    for (int i = 0; i < 16; ++i) {
        int e = t + i * 256;
        int srow = e >> 8;
        float v = bf2f(x[(size_t)bp * 4096 + e]);
        if (padf[bp * Sq + srow]) v = 0.f;
        loc[i] = v; s1 += v; s2 += v * v;
    }
    float mean = blockReduceSum(s1, sh) * (1.0f / 4096.0f);
    float var  = blockReduceSum(s2, sh) * (1.0f / 4096.0f) - mean * mean;
    float inv  = rsqrtf(var + 1e-5f);
    #pragma unroll
    for (int i = 0; i < 16; ++i) {
        int e = t + i * 256;
        out[(size_t)bp * 4096 + e] = f2bf((loc[i] - mean) * inv * g[e] + b[e]);
    }
}

// VQ: per patch, all 4 codebooks; argmin over 1024 codewords of 64-dim dist.
__global__ __launch_bounds__(256) void k_vq(
    const float* __restrict__ lat, const float* __restrict__ cbs,
    const float* __restrict__ pmask, int* __restrict__ ids,
    float* __restrict__ quantf, bf16_t* __restrict__ quantb,
    float* __restrict__ counts, float* __restrict__ acc) {
    const int bp = blockIdx.x;
    const int t  = threadIdx.x;
    __shared__ float ssub[256];
    __shared__ float sd[256];
    __shared__ int   si[256];
    ssub[t] = lat[(size_t)bp * 256 + t];
    __syncthreads();
    for (int c = 0; c < Cq; ++c) {
        const float* sub = &ssub[c * 64];
        float best = 3.4e38f; int bk = 0;
        for (int k = t; k < KCBq; k += 256) {
            const float* cb = cbs + ((size_t)c * KCBq + k) * DSq;
            float d = 0.f;
            for (int j = 0; j < 64; ++j) { float df = sub[j] - cb[j]; d += df * df; }
            if (d < best) { best = d; bk = k; }
        }
        sd[t] = best; si[t] = bk; __syncthreads();
        for (int o = 128; o > 0; o >>= 1) {
            if (t < o && sd[t + o] < sd[t]) { sd[t] = sd[t + o]; si[t] = si[t + o]; }
            __syncthreads();
        }
        const int kwin = si[0];
        float df2 = 0.f;
        if (t < 64) {
            float q = cbs[((size_t)c * KCBq + kwin) * DSq + t];
            quantf[(size_t)bp * 256 + c * 64 + t] = q;
            quantb[(size_t)bp * 256 + c * 64 + t] = f2bf(q);
            float df = sub[t] - q; df2 = df * df;
        }
        float sq = blockReduceSum(df2, sd);
        if (t == 0) {
            atomicAdd(&acc[4], sq);
            ids[bp * Cq + c] = kwin;
            atomicAdd(&counts[c * KCBq + kwin], pmask[bp]);
        }
        __syncthreads();
    }
}

__global__ __launch_bounds__(256) void k_recon(
    const float* __restrict__ logits, const int* __restrict__ pid, float* __restrict__ acc) {
    __shared__ float sh[256];
    const size_t row = blockIdx.x;
    const int t = threadIdx.x;
    const float* lg = logits + row * Vq;
    float a = -1e30f;
    for (int e = t; e < Vq; e += 256) a = fmaxf(a, lg[e]);
    float mx = blockReduceMax(a, sh);
    float s = 0.f;
    for (int e = t; e < Vq; e += 256) s += __expf(lg[e] - mx);
    float sum = blockReduceSum(s, sh);
    if (t == 0) {
        float lse = __logf(sum) + mx;
        int tg = pid[row];
        float keep = (tg != PADq) ? 1.f : 0.f;
        atomicAdd(&acc[0], (lse - lg[tg]) * keep);
        atomicAdd(&acc[1], keep);
    }
}

__global__ __launch_bounds__(256) void k_prior_ln(
    const float* __restrict__ quantf, const float* __restrict__ lng,
    const float* __restrict__ lnb, bf16_t* __restrict__ Hc) {
    __shared__ float sh[256];
    const int j = blockIdx.x;      // 0..M2-1
    const int c = blockIdx.y;      // 0..3
    const int t = threadIdx.x;
    const int b = j / (Pq - 1), p = j % (Pq - 1);
    const float v = quantf[((size_t)(b * Pq + p)) * 256 + t];
    float mean = blockReduceSum(v, sh) * (1.0f / 256.0f);
    float d = v - mean;
    float var = blockReduceSum(d * d, sh) * (1.0f / 256.0f);
    float y = d * rsqrtf(var + 1e-5f) * lng[c * 256 + t] + lnb[c * 256 + t];
    Hc[((size_t)c * M2q + j) * 256 + t] = f2bf(y);
}

__global__ __launch_bounds__(256) void k_prednll(
    const float* __restrict__ lg2, const int* __restrict__ ids,
    const float* __restrict__ pmask, int c, float* __restrict__ acc) {
    __shared__ float sh[256];
    const int j = blockIdx.x;
    const int t = threadIdx.x;
    const float* lg = lg2 + (size_t)j * KCBq;
    float a = -1e30f;
    for (int e = t; e < KCBq; e += 256) a = fmaxf(a, lg[e]);
    float mx = blockReduceMax(a, sh);
    float s = 0.f;
    for (int e = t; e < KCBq; e += 256) s += __expf(lg[e] - mx);
    float sum = blockReduceSum(s, sh);
    if (t == 0) {
        const int b = j / (Pq - 1), p = j % (Pq - 1);
        int tg = ids[((size_t)(b * Pq + p + 1)) * Cq + c];
        float pm = pmask[b * Pq + p] * pmask[b * Pq + p + 1];
        atomicAdd(&acc[2], (__logf(sum) + mx - lg[tg]) * pm);
        if (c == 0) atomicAdd(&acc[3], pm);
    }
}

__global__ void k_init(float* __restrict__ counts, float* __restrict__ acc) {
    int i = blockIdx.x * 256 + threadIdx.x;
    if (i < Cq * KCBq) counts[i] = 0.f;
    if (i < 8) acc[i] = 0.f;
}

__global__ void k_ids(const int* __restrict__ ids, float* __restrict__ out) {
    int i = blockIdx.x * 256 + threadIdx.x;
    if (i < BPq * Cq) out[i] = (float)ids[i];
}

__global__ __launch_bounds__(256) void k_finalize(
    const float* __restrict__ counts, const float* __restrict__ acc,
    const float* __restrict__ pmask, float* __restrict__ out5) {
    __shared__ float sh[256];
    const int t = threadIdx.x;
    float s = 0.f;
    for (int i = t; i < BPq; i += 256) s += pmask[i];
    float msum = fmaxf(blockReduceSum(s, sh), 1.0f);
    float perpsum = 0.f;
    for (int c = 0; c < Cq; ++c) {
        float e = 0.f;
        for (int k = t; k < KCBq; k += 256) {
            float avg = counts[c * KCBq + k] / msum;
            e += avg * __logf(avg + 1e-8f);
        }
        float ent = blockReduceSum(e, sh);
        if (t == 0) perpsum += __expf(-ent);
    }
    if (t == 0) {
        float recon  = acc[0] / fmaxf(acc[1], 1.0f);
        float nel    = (float)BPq * 256.0f;
        float commit = acc[4] / nel * 0.25f;
        float cbl    = acc[4] / nel * 1.0f;
        float pred   = (acc[2] / fmaxf(acc[3], 1.0f)) / (float)Cq * 0.5f;
        out5[0] = recon; out5[1] = commit; out5[2] = cbl; out5[3] = pred;
        out5[4] = perpsum / (float)Cq;
    }
}

// =====================================================================
// host side
// =====================================================================
static void launch_gemm(hipStream_t s, int act,
                        const bf16_t* A, const bf16_t* W, const float* bias,
                        bf16_t* Cb, float* Cf, int M, int N, int K) {
    dim3 g((N + BN - 1) / BN, (M + BM - 1) / BM), b(256);
    if (act == 0)      k_gemm<0><<<g, b, 0, s>>>(A, W, bias, Cb, Cf, M, N, K);
    else if (act == 1) k_gemm<1><<<g, b, 0, s>>>(A, W, bias, Cb, Cf, M, N, K);
    else               k_gemm<2><<<g, b, 0, s>>>(A, W, bias, Cb, Cf, M, N, K);
}

extern "C" void kernel_launch(void* const* d_in, const int* in_sizes, int n_in,
                              void* d_out, int out_size, void* d_ws, size_t ws_size,
                              hipStream_t stream) {
    (void)in_sizes; (void)n_in; (void)out_size; (void)ws_size;

    const int*   patch_ids  = (const int*)  d_in[0];
    const float* patch_mask = (const float*)d_in[1];
    const float* byte_emb   = (const float*)d_in[2];
    const float* enc_pos    = (const float*)d_in[3];
    const float* dec_pos    = (const float*)d_in[4];
    const float* enc_p[12]; for (int i = 0; i < 12; ++i) enc_p[i] = (const float*)d_in[5 + i];
    const float* dec_p[12]; for (int i = 0; i < 12; ++i) dec_p[i] = (const float*)d_in[17 + i];
    const float* tl_lng = (const float*)d_in[29];
    const float* tl_lnb = (const float*)d_in[30];
    const float* tl_w1  = (const float*)d_in[31];
    const float* tl_b1  = (const float*)d_in[32];
    const float* tl_w2  = (const float*)d_in[33];
    const float* tl_b2  = (const float*)d_in[34];
    const float* ex_w   = (const float*)d_in[35];
    const float* ex_b   = (const float*)d_in[36];
    const float* out_w  = (const float*)d_in[37];
    const float* out_b  = (const float*)d_in[38];
    const float* codebooks = (const float*)d_in[39];
    const float* pr_lng = (const float*)d_in[40];
    const float* pr_lnb = (const float*)d_in[41];
    const float* pr_w1  = (const float*)d_in[42];
    const float* pr_b1  = (const float*)d_in[43];
    const float* pr_w2  = (const float*)d_in[44];
    const float* pr_b2  = (const float*)d_in[45];

    char* wsp = (char*)d_ws; size_t off = 0;
    auto balloc = [&](size_t bytes) -> void* {
        void* p = wsp + off;
        off += (bytes + 255) & ~(size_t)255;
        return p;
    };
    auto cast_w = [&](const float* src, size_t n) -> bf16_t* {
        bf16_t* dst = (bf16_t*)balloc(n * sizeof(bf16_t));
        size_t grid = (n + 255) / 256; if (grid > 8192) grid = 8192;
        k_cast_bf16<<<dim3((unsigned)grid), dim3(256), 0, stream>>>(src, dst, n);
        return dst;
    };

    // --- weight conversions (f32 -> bf16, whole stacked tensors) ---
    bf16_t* enc_wqkv = cast_w(enc_p[0], (size_t)Lq * 3 * Eq * Eq);
    bf16_t* enc_wo   = cast_w(enc_p[2], (size_t)Lq * Eq * Eq);
    bf16_t* enc_w1   = cast_w(enc_p[6], (size_t)Lq * 4 * Eq * Eq);
    bf16_t* enc_w2   = cast_w(enc_p[8], (size_t)Lq * 4 * Eq * Eq);
    bf16_t* dec_wqkv = cast_w(dec_p[0], (size_t)Lq * 3 * Eq * Eq);
    bf16_t* dec_wo   = cast_w(dec_p[2], (size_t)Lq * Eq * Eq);
    bf16_t* dec_w1   = cast_w(dec_p[6], (size_t)Lq * 4 * Eq * Eq);
    bf16_t* dec_w2   = cast_w(dec_p[8], (size_t)Lq * 4 * Eq * Eq);
    bf16_t* tlw1 = cast_w(tl_w1, (size_t)2 * DLATq * Sq * Eq);
    bf16_t* tlw2 = cast_w(tl_w2, (size_t)DLATq * 2 * DLATq);
    bf16_t* exw  = cast_w(ex_w,  (size_t)Eq * Sq * DLATq);
    bf16_t* outw = cast_w(out_w, (size_t)Vq * Eq);
    bf16_t* prw1 = cast_w(pr_w1, (size_t)Cq * DLATq * DLATq);
    bf16_t* prw2 = cast_w(pr_w2, (size_t)Cq * KCBq * DLATq);

    // --- activation / scratch buffers ---
    bf16_t* xb    = (bf16_t*)balloc((size_t)MTOK * Eq * 2);
    bf16_t* qkvb  = (bf16_t*)balloc((size_t)MTOK * 3 * Eq * 2);
    bf16_t* ab    = (bf16_t*)balloc((size_t)MTOK * Eq * 2);
    bf16_t* fb    = (bf16_t*)balloc((size_t)MTOK * 4 * Eq * 2);
    bf16_t* tbuf  = (bf16_t*)balloc((size_t)MTOK * Eq * 2);
    unsigned char* padf = (unsigned char*)balloc(MTOK);
    bf16_t* h4096 = (bf16_t*)balloc((size_t)BPq * 4096 * 2);
    bf16_t* t512  = (bf16_t*)balloc((size_t)BPq * 512 * 2);
    float*  latf  = (float*)balloc((size_t)BPq * 256 * 4);
    float*  quantf= (float*)balloc((size_t)BPq * 256 * 4);
    bf16_t* quantb= (bf16_t*)balloc((size_t)BPq * 256 * 2);
    int*    idsi  = (int*)balloc((size_t)BPq * Cq * 4);
    float*  counts= (float*)balloc((size_t)Cq * KCBq * 4);
    float*  acc   = (float*)balloc(64);
    bf16_t* Hc    = (bf16_t*)balloc((size_t)Cq * M2q * 256 * 2);
    bf16_t* Gb    = (bf16_t*)balloc((size_t)M2q * 256 * 2);
    float*  plog  = (float*)balloc((size_t)M2q * KCBq * 4);

    k_init<<<(Cq * KCBq + 255) / 256, 256, 0, stream>>>(counts, acc);

    // --- encoder input: embed + pos, pad flags ---
    k_embed<<<(MTOK * Eq) / 256, 256, 0, stream>>>(patch_ids, byte_emb, enc_pos, xb, padf);

    auto tlayer = [&](const float* const* p, bf16_t* wqkv_a, bf16_t* wo_a,
                      bf16_t* w1_a, bf16_t* w2_a, int l, int useMask) {
        launch_gemm(stream, 0, xb, wqkv_a + (size_t)l * 3 * Eq * Eq,
                    p[1] + (size_t)l * 3 * Eq, qkvb, nullptr, MTOK, 3 * Eq, Eq);
        k_attn<<<BPq * Hq, 32, 0, stream>>>(qkvb, padf, ab, useMask);
        launch_gemm(stream, 0, ab, wo_a + (size_t)l * Eq * Eq,
                    p[3] + (size_t)l * Eq, tbuf, nullptr, MTOK, Eq, Eq);
        k_ln256<<<MTOK, 256, 0, stream>>>(tbuf, xb, p[4] + (size_t)l * Eq, p[5] + (size_t)l * Eq, xb);
        launch_gemm(stream, 1, xb, w1_a + (size_t)l * 4 * Eq * Eq,
                    p[7] + (size_t)l * 4 * Eq, fb, nullptr, MTOK, 4 * Eq, Eq);
        launch_gemm(stream, 0, fb, w2_a + (size_t)l * 4 * Eq * Eq,
                    p[9] + (size_t)l * Eq, tbuf, nullptr, MTOK, Eq, 4 * Eq);
        k_ln256<<<MTOK, 256, 0, stream>>>(tbuf, xb, p[10] + (size_t)l * Eq, p[11] + (size_t)l * Eq, xb);
    };

    for (int l = 0; l < Lq; ++l) tlayer(enc_p, enc_wqkv, enc_wo, enc_w1, enc_w2, l, 1);

    // --- latent head: mask + LN(4096) -> gelu GEMM -> GEMM ---
    k_maskln<<<BPq, 256, 0, stream>>>(xb, padf, tl_lng, tl_lnb, h4096);
    launch_gemm(stream, 2, h4096, tlw1, tl_b1, t512, nullptr, BPq, 2 * DLATq, Sq * Eq);
    launch_gemm(stream, 0, t512, tlw2, tl_b2, nullptr, latf, BPq, DLATq, 2 * DLATq);

    // --- VQ ---
    k_vq<<<BPq, 256, 0, stream>>>(latf, codebooks, patch_mask, idsi, quantf, quantb, counts, acc);

    // --- decoder: expand + pos, 4 layers, no mask ---
    launch_gemm(stream, 2, quantb, exw, ex_b, h4096, nullptr, BPq, Sq * Eq, DLATq);
    k_addpos<<<(MTOK * Eq) / 256, 256, 0, stream>>>(h4096, dec_pos, xb);
    for (int l = 0; l < Lq; ++l) tlayer(dec_p, dec_wqkv, dec_wo, dec_w1, dec_w2, l, 0);

    // --- logits (straight to d_out, f32) + recon loss ---
    float* outf = (float*)d_out;
    launch_gemm(stream, 0, xb, outw, out_b, nullptr, outf, MTOK, Vq, Eq);
    k_recon<<<MTOK, 256, 0, stream>>>(outf, patch_ids, acc);

    // --- prior heads (per codebook) ---
    k_prior_ln<<<dim3(M2q, Cq), 256, 0, stream>>>(quantf, pr_lng, pr_lnb, Hc);
    for (int c = 0; c < Cq; ++c) {
        launch_gemm(stream, 2, Hc + (size_t)c * M2q * 256, prw1 + (size_t)c * DLATq * DLATq,
                    pr_b1 + (size_t)c * DLATq, Gb, nullptr, M2q, DLATq, DLATq);
        launch_gemm(stream, 0, Gb, prw2 + (size_t)c * KCBq * DLATq,
                    pr_b2 + (size_t)c * KCBq, nullptr, plog, M2q, KCBq, DLATq);
        k_prednll<<<M2q, 256, 0, stream>>>(plog, idsi, patch_mask, c, acc);
    }

    // --- outputs: ids (as float) + scalars ---
    k_ids<<<(BPq * Cq + 255) / 256, 256, 0, stream>>>(idsi, outf + (size_t)MTOK * Vq);
    k_finalize<<<1, 256, 0, stream>>>(counts, acc, patch_mask,
                                      outf + (size_t)MTOK * Vq + (size_t)BPq * Cq);
}
